// MultiHeadAttention_86423331930508
// MI455X (gfx1250) — matrix operand order repaired and checked by reading, not compiled, not run on hardware
//
#include <hip/hip_runtime.h>
#include <hip/hip_bf16.h>

// ---------------------------------------------------------------------------
// CDNA5 (gfx1250) fused transformer layer, bf16 WMMA + TDM staging:
//   0) one fp32->bf16 pass for all GEMM operands.
//   1) QKV projections (bf16 WMMA): LDS tiles staged by the Tensor Data Mover
//      (tensor_load_to_lds, TENSORcnt), D# pad fields implement the padded
//      LDS pitch in hardware.
//   2) fused attention: scores -> softmax -> single write of the 512MB attn
//      output -> P@V, probabilities kept bf16 in LDS, V stored transposed.
//   3) LN -> FFN (bf16 WMMA) -> LN.
// ---------------------------------------------------------------------------

#ifndef TDM_STAGING
#define TDM_STAGING 1   // 0 = manual b128 copy fallback
#endif

typedef unsigned short u16;
typedef __attribute__((ext_vector_type(16))) __bf16    v16bf;
typedef __attribute__((ext_vector_type(8)))  float     v8f;
typedef __attribute__((ext_vector_type(4)))  unsigned  v4u;
typedef __attribute__((ext_vector_type(8)))  unsigned  v8u;

union BFrag {
    uint4    q[2];   // two 16B chunks (16 bf16 = 32B)
    unsigned u[8];
    v16bf    v;
};

__device__ __forceinline__ u16 f2bf(float f) {
    union { float f; unsigned u; } x; x.f = f;
    unsigned u = x.u;
    u += 0x7fffu + ((u >> 16) & 1u);   // round-to-nearest-even
    return (u16)(u >> 16);
}
__device__ __forceinline__ unsigned pk2(float a, float b) {
    return (unsigned)f2bf(a) | ((unsigned)f2bf(b) << 16);
}

// ---------------------------------------------------------------------------
// Tensor Data Mover: 2D tile (8-byte units) global -> LDS.  Builds the D#
// (group0: 4 SGPRs, group1: 8 SGPRs) per CDNA5 ISA and
// issues the 2-group form of tensor_load_to_lds (VADDR2/3 = NULL -> <=2D).
// Caller: one wave only; follow with s_wait_tensorcnt(0) + barrier.
//   pad fields: insert pad_amount dwords every pad_interval dwords (load only)
// ---------------------------------------------------------------------------
__device__ __forceinline__ void tdm_load_2d(
    unsigned lds_off, const void* gptr,
    unsigned tensor_w_units, unsigned tensor_h,
    unsigned tile_w_units, unsigned tile_h,
    unsigned row_stride_units, unsigned pad_fields)
{
    const unsigned long long ga = (unsigned long long)(size_t)gptr;
    v4u g0;
    g0.x = 1u;                                    // count=1, user descriptor
    g0.y = lds_off;                               // lds_addr (bytes)
    g0.z = (unsigned)(ga & 0xffffffffu);          // global_addr[31:0]
    g0.w = (unsigned)((ga >> 32) & 0x1ffffffu)    // global_addr[56:32]
         | (2u << 30);                            // type = 2 ("image")
    v8u g1;
    g1[0] = (3u << 16) | pad_fields;              // data_size=8B, pad ctrl
    g1[1] = (tensor_w_units & 0xffffu) << 16;     // tensor_dim0[15:0]
    g1[2] = ((tensor_w_units >> 16) & 0xffffu)    // tensor_dim0[31:16]
          | ((tensor_h & 0xffffu) << 16);         // tensor_dim1[15:0]
    g1[3] = ((tensor_h >> 16) & 0xffffu)          // tensor_dim1[31:16]
          | ((tile_w_units & 0xffffu) << 16);     // tile_dim0
    g1[4] = (tile_h & 0xffffu);                   // tile_dim1 (tile_dim2=0)
    g1[5] = row_stride_units;                     // tensor_dim0_stride[31:0]
    g1[6] = 0u;                                   // stride hi + dim1_stride lo
    g1[7] = 0u;                                   // dim1_stride hi
    asm volatile("tensor_load_to_lds %0, %1" :: "s"(g0), "s"(g1) : "memory");
}

// pad: every 32 dwords (64 bf16 line) insert 4 dwords -> 72-u16 LDS pitch
#define TDM_PAD_72 ((1u << 20) | (4u << 22) | (3u << 25))

// ---------------------------------------------------------------------------
// fp32 -> bf16 elementwise (8 elements / thread, b128 in, b128 out)
// ---------------------------------------------------------------------------
__global__ __launch_bounds__(256) void convert_bf16_kernel(
    const float* __restrict__ src, u16* __restrict__ dst, int n8)
{
    const int idx = blockIdx.x * 256 + threadIdx.x;
    if (idx >= n8) return;
    const float4 a = ((const float4*)src)[idx * 2];
    const float4 b = ((const float4*)src)[idx * 2 + 1];
    uint4 o;
    o.x = pk2(a.x, a.y);
    o.y = pk2(a.z, a.w);
    o.z = pk2(b.x, b.y);
    o.w = pk2(b.z, b.w);
    ((uint4*)dst)[idx] = o;
}

// ---------------------------------------------------------------------------
// GEMM: Y[m,n] = act( A[m,:] . W[n,:] + bias[n] ),  A:[M,K] bf16, W:[N,K] bf16
// Block tile 128x64, K staged 64-deep in LDS (TDM or b128 copies).
// 8 waves x (2x2) 16x16 WMMA tiles.
// mode: 0 = fp32 [M,N]; 1 = bf16 head-major [h*B+b][l][64];
//       2 = bf16 head-major transposed [h*B+b][64][L]; 3 = bf16 [M,N]
// ---------------------------------------------------------------------------
#define GK_BM   128
#define GK_BN   64
#define GK_BK   64
#define GK_KSTR 72   // u16 pitch; 144B keeps 16B alignment; matches TDM_PAD_72

__global__ __launch_bounds__(256) void wmma_gemm_kernel(
    const u16* __restrict__ A, const u16* __restrict__ W,
    const float* __restrict__ bias, float* __restrict__ Yf,
    u16* __restrict__ Yb, int K, int N, int relu, int mode, int Bb, int L)
{
    __shared__ __align__(16) u16 As[GK_BM * GK_KSTR];
    __shared__ __align__(16) u16 Bs[GK_BN * GK_KSTR];

    const int tid  = threadIdx.x;
    const int lane = tid & 31;
    const int wid  = tid >> 5;
    const int wm   = wid & 3;        // 4 wave-rows of 32
    const int wn   = wid >> 2;       // 2 wave-cols of 32
    const int h    = lane >> 4;
    const int l16  = lane & 15;
    const int rowBase = blockIdx.x * GK_BM;
    const int colBase = blockIdx.y * GK_BN;

    v8f acc[2][2];
    #pragma unroll
    for (int i = 0; i < 2; i++)
        #pragma unroll
        for (int j = 0; j < 2; j++)
            acc[i][j] = (v8f){0.f,0.f,0.f,0.f,0.f,0.f,0.f,0.f};

    for (int k0 = 0; k0 < K; k0 += GK_BK) {
#if TDM_STAGING
        if (wid == 0) {
            // A tile: 128 lines x 64 bf16 (16 x 8B units), pitch K/4 units
            tdm_load_2d((unsigned)(size_t)(void*)As,
                        &A[(size_t)rowBase * K + k0],
                        (unsigned)(K >> 2), 1u << 20,
                        GK_BK / 4, GK_BM, (unsigned)(K >> 2), TDM_PAD_72);
            // W tile: 64 lines x 64 bf16
            tdm_load_2d((unsigned)(size_t)(void*)Bs,
                        &W[(size_t)colBase * K + k0],
                        (unsigned)(K >> 2), 1u << 20,
                        GK_BK / 4, GK_BN, (unsigned)(K >> 2), TDM_PAD_72);
            __builtin_amdgcn_s_wait_tensorcnt(0);
        }
#else
        {   // manual path: issue all loads, then all LDS stores
            uint4 ta[4], tb[2];
            #pragma unroll
            for (int i = 0; i < 4; i++) {
                int idx = tid + i * 256;
                int r = idx >> 3, c8 = (idx & 7) * 8;
                ta[i] = *(const uint4*)&A[(size_t)(rowBase + r) * K + k0 + c8];
            }
            #pragma unroll
            for (int i = 0; i < 2; i++) {
                int idx = tid + i * 256;
                int r = idx >> 3, c8 = (idx & 7) * 8;
                tb[i] = *(const uint4*)&W[(size_t)(colBase + r) * K + k0 + c8];
            }
            #pragma unroll
            for (int i = 0; i < 4; i++) {
                int idx = tid + i * 256;
                int r = idx >> 3, c8 = (idx & 7) * 8;
                *(uint4*)&As[r * GK_KSTR + c8] = ta[i];
            }
            #pragma unroll
            for (int i = 0; i < 2; i++) {
                int idx = tid + i * 256;
                int r = idx >> 3, c8 = (idx & 7) * 8;
                *(uint4*)&Bs[r * GK_KSTR + c8] = tb[i];
            }
        }
#endif
        __syncthreads();

        #pragma unroll
        for (int ks = 0; ks < GK_BK / 32; ks++) {
            const int kb = ks * 32;
            BFrag a[2], b[2];
            #pragma unroll
            for (int i = 0; i < 2; i++) {
                const u16* pa = &As[(wm * 32 + i * 16 + l16) * GK_KSTR + kb + 8 * h];
                a[i].q[0] = *(const uint4*)pa;
                a[i].q[1] = *(const uint4*)(pa + 16);
            }
            #pragma unroll
            for (int j = 0; j < 2; j++) {
                const u16* pb = &Bs[(wn * 32 + j * 16 + l16) * GK_KSTR + kb + 8 * h];
                b[j].q[0] = *(const uint4*)pb;
                b[j].q[1] = *(const uint4*)(pb + 16);
            }
            #pragma unroll
            for (int i = 0; i < 2; i++)
                #pragma unroll
                for (int j = 0; j < 2; j++)
                    acc[i][j] = __builtin_amdgcn_wmma_f32_16x16x32_bf16(
                        false, a[i].v, false, b[j].v, (short)0, acc[i][j],
                        false, false);
        }
        __syncthreads();
    }

    // --- epilogue ---
    #pragma unroll
    for (int i = 0; i < 2; i++) {
        #pragma unroll
        for (int j = 0; j < 2; j++) {
            const int gn = colBase + wn * 32 + j * 16 + l16;
            const float bv = bias ? bias[gn] : 0.0f;
            #pragma unroll
            for (int r = 0; r < 8; r++) {
                const int gm = rowBase + wm * 32 + i * 16 + 8 * h + r;
                float v = acc[i][j][r] + bv;
                if (relu) v = fmaxf(v, 0.0f);
                if (mode == 0) {
                    Yf[(size_t)gm * N + gn] = v;
                } else if (mode == 3) {
                    Yb[(size_t)gm * N + gn] = f2bf(v);
                } else {
                    const int bb = gm / L, ll = gm % L;
                    const int hd = gn >> 6, d = gn & 63;
                    if (mode == 1)
                        Yb[(((size_t)hd * Bb + bb) * L + ll) * 64 + d] = f2bf(v);
                    else
                        Yb[(((size_t)hd * Bb + bb) * 64 + d) * L + ll] = f2bf(v);
                }
            }
        }
    }
}

// ---------------------------------------------------------------------------
// Fused attention. block = (16-row q tile, bh = h*B+b). bf16 Q/K head-major,
// bf16 V transposed. LDS: fp32 scores 16xL + bf16 probs 16xL + Q + reduce +
// partial-O (~204KB of the 320KB WGP LDS, dynamic).
// ---------------------------------------------------------------------------
__global__ __launch_bounds__(256) void attention_kernel(
    const u16* __restrict__ Qh, const u16* __restrict__ Kh,
    const u16* __restrict__ Vt, const int* __restrict__ maskp,
    float* __restrict__ attn_out, float* __restrict__ ctx_out,
    int B, int L)
{
    extern __shared__ __align__(16) char smem[];
    const int Lk = L;
    float* S   = (float*)smem;                                   // 16*Lk f32
    u16*   Pb  = (u16*)(smem + (size_t)16 * Lk * 4);             // 16*Lk bf16
    u16*   Qs  = (u16*)((char*)Pb + (size_t)16 * Lk * 2);        // 16*64 bf16
    float* red = (float*)((char*)Qs + 16 * 64 * 2);              // 16*16 f32
    float* Op  = red + 256;                                      // 8*16*16 f32

    const int tid  = threadIdx.x;
    const int lane = tid & 31;
    const int wid  = tid >> 5;
    const int h    = lane >> 4;
    const int l16  = lane & 15;
    const int qt   = blockIdx.x;
    const int bh   = blockIdx.y;
    const float scale = 1.0f / (8.0f + 1e-6f);   // 1/(sqrt(64)+1e-6)

    // stage Q tile (16x64 bf16, contiguous 2KB)
#if TDM_STAGING
    if (wid == 0) {
        tdm_load_2d((unsigned)(size_t)(void*)Qs,
                    Qh + (((size_t)bh * L) + qt * 16) * 64,
                    256u, 1u, 256u, 1u, 256u, 0u);
        __builtin_amdgcn_s_wait_tensorcnt(0);
    }
#else
    if (tid < 128)
        ((uint4*)Qs)[tid] =
            ((const uint4*)(Qh + (((size_t)bh * L) + qt * 16) * 64))[tid];
#endif
    __syncthreads();

    BFrag a0, a1;
    {
        const u16* p = &Qs[l16 * 64 + 8 * h];
        a0.q[0] = *(const uint4*)p;  a0.q[1] = *(const uint4*)(p + 16);
        const u16* p2 = &Qs[l16 * 64 + 32 + 8 * h];
        a1.q[0] = *(const uint4*)p2; a1.q[1] = *(const uint4*)(p2 + 16);
    }

    // pad mask: batch-major fold (bh / N_HEAD) masks whole query rows
    const int pad_b = bh >> 4;   // N_HEAD == 16
    int rowpad = 0;
    #pragma unroll
    for (int r = 0; r < 8; r++) {
        int qg = qt * 16 + 8 * h + r;
        if (maskp[pad_b * L + qg] != 0) rowpad |= (1 << r);
    }

    // ---- Phase 1: scores ----
    for (int nt = wid; nt < Lk / 16; nt += 8) {
        const u16* kb = Kh + (((size_t)bh * L) + nt * 16 + l16) * 64;
        BFrag b0, b1;
        b0.q[0] = *(const uint4*)(kb + 8 * h);
        b0.q[1] = *(const uint4*)(kb + 8 * h + 16);
        b1.q[0] = *(const uint4*)(kb + 32 + 8 * h);
        b1.q[1] = *(const uint4*)(kb + 32 + 8 * h + 16);
        v8f c = (v8f){0.f,0.f,0.f,0.f,0.f,0.f,0.f,0.f};
        c = __builtin_amdgcn_wmma_f32_16x16x32_bf16(false, a0.v, false, b0.v,
                                                    (short)0, c, false, false);
        c = __builtin_amdgcn_wmma_f32_16x16x32_bf16(false, a1.v, false, b1.v,
                                                    (short)0, c, false, false);
        #pragma unroll
        for (int r = 0; r < 8; r++) {
            int m  = 8 * h + r;
            int qg = qt * 16 + m;
            int ng = nt * 16 + l16;
            float v = c[r] * scale;
            if (ng > qg || ((rowpad >> r) & 1)) v = -1e10f;
            S[(size_t)m * Lk + ng] = v;
        }
    }
    __syncthreads();

    // ---- Phase 2: softmax, write attn (fp32), keep probs bf16 in LDS ----
    {
        const int row = tid >> 4, c16 = tid & 15;
        float mx = -3.4e38f;
        for (int j = c16; j < Lk; j += 16) mx = fmaxf(mx, S[(size_t)row * Lk + j]);
        red[row * 16 + c16] = mx;
        __syncthreads();
        mx = red[row * 16];
        #pragma unroll
        for (int t = 1; t < 16; t++) mx = fmaxf(mx, red[row * 16 + t]);
        float sum = 0.f;
        for (int j = c16; j < Lk; j += 16) {
            float e = __expf(S[(size_t)row * Lk + j] - mx);
            S[(size_t)row * Lk + j] = e;
            sum += e;
        }
        __syncthreads();
        red[row * 16 + c16] = sum;
        __syncthreads();
        float s = 0.f;
        #pragma unroll
        for (int t = 0; t < 16; t++) s += red[row * 16 + t];
        const float inv = 1.0f / s;
        const int qg = qt * 16 + row;
        float* arow = attn_out + ((size_t)bh * L + qg) * Lk;
        for (int j = c16; j < Lk; j += 16) {
            float p = S[(size_t)row * Lk + j] * inv;
            arow[j] = p;                          // mandatory attn output
            Pb[(size_t)row * Lk + j] = f2bf(p);   // bf16 probs for P@V
        }
    }
    __syncthreads();

    // ---- Phase 3: O = P @ V (waves: 4 col-tiles x 2 K-halves) ----
    {
        const int ntile = wid & 3;
        const int khalf = wid >> 2;
        const int col   = ntile * 16 + l16;
        v8f c = (v8f){0.f,0.f,0.f,0.f,0.f,0.f,0.f,0.f};
        for (int ks = 0; ks < Lk / 64; ks++) {
            const int k0 = khalf * (Lk / 2) + ks * 32;
            BFrag a, b;
            const u16* pa = &Pb[(size_t)l16 * Lk + k0 + 8 * h];
            a.q[0] = *(const uint4*)pa;
            a.q[1] = *(const uint4*)(pa + 16);
            const u16* pv = &Vt[((size_t)bh * 64 + col) * L + k0 + 8 * h];
            b.q[0] = *(const uint4*)pv;
            b.q[1] = *(const uint4*)(pv + 16);
            c = __builtin_amdgcn_wmma_f32_16x16x32_bf16(false, a.v, false, b.v,
                                                        (short)0, c, false, false);
        }
        #pragma unroll
        for (int r = 0; r < 8; r++) {
            int m = 8 * h + r;
            Op[((khalf * 4 + ntile) * 16 + m) * 16 + l16] = c[r];
        }
    }
    __syncthreads();
    for (int i = tid; i < 16 * 64; i += 256) {
        const int m = i >> 6, col = i & 63;
        const int ntile = col >> 4, nl = col & 15;
        float v = Op[((ntile) * 16 + m) * 16 + nl] +
                  Op[((4 + ntile) * 16 + m) * 16 + nl];
        const int bb = bh % B, hh = bh / B;   // head-major bh = h*B + b
        const int qg = qt * 16 + m;
        ctx_out[(((size_t)bb * L) + qg) * 1024 + hh * 64 + col] = v;
    }
}

// ---------------------------------------------------------------------------
// LayerNorm: out = LN(a + b) * g + beta (eps 1e-5); optional bf16 copy.
// ---------------------------------------------------------------------------
__global__ __launch_bounds__(256) void layernorm_kernel(
    const float* __restrict__ a, const float* __restrict__ b,
    const float* __restrict__ g, const float* __restrict__ beta,
    float* __restrict__ out, u16* __restrict__ outb)
{
    __shared__ float red[256];
    const int row = blockIdx.x, tid = threadIdx.x;
    float x[4];
    float s = 0.f;
    #pragma unroll
    for (int i = 0; i < 4; i++) {
        const int c = tid + i * 256;
        x[i] = a[(size_t)row * 1024 + c] + b[(size_t)row * 1024 + c];
        s += x[i];
    }
    red[tid] = s; __syncthreads();
    for (int off = 128; off > 0; off >>= 1) {
        if (tid < off) red[tid] += red[tid + off];
        __syncthreads();
    }
    const float mu = red[0] * (1.0f / 1024.0f);
    __syncthreads();
    s = 0.f;
    #pragma unroll
    for (int i = 0; i < 4; i++) { float d = x[i] - mu; s += d * d; }
    red[tid] = s; __syncthreads();
    for (int off = 128; off > 0; off >>= 1) {
        if (tid < off) red[tid] += red[tid + off];
        __syncthreads();
    }
    const float rinv = rsqrtf(red[0] * (1.0f / 1024.0f) + 1e-5f);
    #pragma unroll
    for (int i = 0; i < 4; i++) {
        const int c = tid + i * 256;
        const float v = (x[i] - mu) * rinv * g[c] + beta[c];
        out[(size_t)row * 1024 + c] = v;
        if (outb) outb[(size_t)row * 1024 + c] = f2bf(v);
    }
}

// ---------------------------------------------------------------------------
extern "C" void kernel_launch(void* const* d_in, const int* in_sizes, int n_in,
                              void* d_out, int out_size, void* d_ws, size_t ws_size,
                              hipStream_t stream)
{
    const float* q    = (const float*)d_in[0];
    const float* k    = (const float*)d_in[1];
    const float* v    = (const float*)d_in[2];
    const int*   mask = (const int*)  d_in[3];
    const float* w_qs = (const float*)d_in[4];
    const float* w_ks = (const float*)d_in[5];
    const float* w_vs = (const float*)d_in[6];
    const float* w1   = (const float*)d_in[7];
    const float* b1   = (const float*)d_in[8];
    const float* w2   = (const float*)d_in[9];
    const float* b2   = (const float*)d_in[10];
    const float* ln_g = (const float*)d_in[11];
    const float* ln_b = (const float*)d_in[12];

    const int B = 2, L = 2048, DM = 1024, BH = 32;
    const size_t TOK = (size_t)B * L;      // 4096 rows
    const size_t MAT = TOK * DM;           // 4,194,304 elements
    const size_t WMAT = (size_t)DM * DM;   // 1,048,576 elements

    // ---- workspace layout: fp32 region first, then bf16 (u16) region ----
    float* ao = (float*)d_ws;              // attention context [B,L,1024]
    float* X  = ao + MAT;                  // LN1 output (residual for LN2)
    float* Fb = X  + MAT;                  // FFN2 output
    u16* u = (u16*)(Fb + MAT);
    u16* qb  = u;            u += MAT;     // bf16 activations
    u16* kb  = u;            u += MAT;
    u16* vb  = u;            u += MAT;
    u16* wqb = u;            u += WMAT;    // bf16 weights
    u16* wkb = u;            u += WMAT;
    u16* wvb = u;            u += WMAT;
    u16* w1b = u;            u += WMAT;
    u16* w2b = u;            u += WMAT;
    u16* qhb = u;            u += MAT;     // head-major Q [bh][l][64]
    u16* khb = u;            u += MAT;     // head-major K
    u16* vtb = u;            u += MAT;     // transposed V [bh][64][L]
    u16* Xb  = u;            u += MAT;     // bf16 LN1 output
    u16* Hbb = u;            u += MAT;     // bf16 FFN hidden

    float* out2 = (float*)d_out;           // [B,L,1024]
    float* attn = out2 + MAT;              // [BH,L,L]

    const dim3 blk(256);

    // ---- 0: one-shot bf16 conversion of all GEMM operands ----
    const int a8 = (int)(MAT / 8), w8 = (int)(WMAT / 8);
    convert_bf16_kernel<<<dim3((a8 + 255) / 256), blk, 0, stream>>>(q, qb, a8);
    convert_bf16_kernel<<<dim3((a8 + 255) / 256), blk, 0, stream>>>(k, kb, a8);
    convert_bf16_kernel<<<dim3((a8 + 255) / 256), blk, 0, stream>>>(v, vb, a8);
    convert_bf16_kernel<<<dim3((w8 + 255) / 256), blk, 0, stream>>>(w_qs, wqb, w8);
    convert_bf16_kernel<<<dim3((w8 + 255) / 256), blk, 0, stream>>>(w_ks, wkb, w8);
    convert_bf16_kernel<<<dim3((w8 + 255) / 256), blk, 0, stream>>>(w_vs, wvb, w8);
    convert_bf16_kernel<<<dim3((w8 + 255) / 256), blk, 0, stream>>>(w1, w1b, w8);
    convert_bf16_kernel<<<dim3((w8 + 255) / 256), blk, 0, stream>>>(w2, w2b, w8);

    const dim3 gg(TOK / GK_BM, DM / GK_BN);

    // ---- 1: QKV projections ----
    wmma_gemm_kernel<<<gg, blk, 0, stream>>>(qb, wqb, nullptr, nullptr, qhb,
                                             DM, DM, 0, 1, B, L);
    wmma_gemm_kernel<<<gg, blk, 0, stream>>>(kb, wkb, nullptr, nullptr, khb,
                                             DM, DM, 0, 1, B, L);
    wmma_gemm_kernel<<<gg, blk, 0, stream>>>(vb, wvb, nullptr, nullptr, vtb,
                                             DM, DM, 0, 2, B, L);

    // ---- 2: fused attention ----
    const size_t shmem = (size_t)16 * L * 4   // S fp32
                       + (size_t)16 * L * 2   // Pb bf16
                       + 16 * 64 * 2          // Qs
                       + 256 * 4              // red
                       + 8 * 16 * 16 * 4;     // Op
    attention_kernel<<<dim3(L / 16, BH), blk, shmem, stream>>>(
        qhb, khb, vtb, mask, attn, ao, B, L);

    // ---- 3: LN + FFN + LN ----
    layernorm_kernel<<<dim3((unsigned)TOK), blk, 0, stream>>>(ao, q, ln_g, ln_b, X, Xb);
    wmma_gemm_kernel<<<gg, blk, 0, stream>>>(Xb, w1b, b1, nullptr, Hbb,
                                             DM, DM, 1, 3, B, L);
    wmma_gemm_kernel<<<gg, blk, 0, stream>>>(Hbb, w2b, b2, Fb, nullptr,
                                             DM, DM, 0, 0, B, L);
    layernorm_kernel<<<dim3((unsigned)TOK), blk, 0, stream>>>(Fb, X, ln_g, ln_b,
                                                              out2, nullptr);
}
